// SpatialTransformer_60825326846281
// MI455X (gfx1250) — matrix operand order repaired and checked by reading, not compiled, not run on hardware
//
#include <hip/hip_runtime.h>
#include <hip/hip_bf16.h>
#include <math.h>

// ---------------------------------------------------------------------------
// Types for CDNA5 WMMA (gfx1250, wave32)
// ---------------------------------------------------------------------------
typedef __attribute__((ext_vector_type(16))) __bf16          v16bf;
typedef __attribute__((ext_vector_type(8)))  float           v8f;
typedef __attribute__((ext_vector_type(8)))  unsigned short  u16x8;
typedef __attribute__((ext_vector_type(16))) unsigned short  u16x16;

typedef unsigned short ushort_t;

__device__ __forceinline__ unsigned short f32_to_bf16(float f) {
  unsigned int u = __builtin_bit_cast(unsigned int, f);
  unsigned int r = u + 0x7fffu + ((u >> 16) & 1u);   // round-to-nearest-even
  return (unsigned short)(r >> 16);
}
__device__ __forceinline__ float bf16_to_f32(unsigned short b) {
  return __builtin_bit_cast(float, (unsigned int)b << 16);
}

struct AFragBits { u16x8 lo, hi; };

// A-matrix fragment (16x32 bf16, MxK), contiguous k order used by this code (not the ISA table order):
//  lane<16 : row m=lane, halves 0..15 -> K 0..15
//  lane>=16: row m=lane-16, halves 0..15 -> K 16..31
__device__ __forceinline__ v16bf load_afrag(const ushort_t* rowptr, int half) {
  AFragBits t;
  t.lo = *(const u16x8*)(rowptr + (half ? 16 : 0));
  t.hi = *(const u16x8*)(rowptr + (half ? 24 : 8));
  return __builtin_bit_cast(v16bf, t);
}
// B-matrix fragment (32x16 bf16, KxN): lane holds one column; lanes 0..15 hold
// K 0..15 contiguously, lanes 16..31 hold K 16..31. p already offset by half*16.
__device__ __forceinline__ v16bf load_bfrag(const ushort_t* p) {
  u16x16 t = *(const u16x16*)p;
  return __builtin_bit_cast(v16bf, t);
}
__device__ __forceinline__ v8f wmma_bf16(v16bf a, v16bf b, v8f c) {
  return __builtin_amdgcn_wmma_f32_16x16x32_bf16(false, a, false, b,
                                                 (short)0, c, false, false);
}

// ---------------------------------------------------------------------------
// Weight convert: produce Bt[N][K] bf16.
//  srcNK==1 : src already [N][K]   (conv weights: [out,in]) or plain flatten
//  srcNK==0 : src is [K][N]        (attention/FF weights stored (in,out))
// ---------------------------------------------------------------------------
__global__ void wconv_kernel(const float* __restrict__ src, ushort_t* __restrict__ dst,
                             int N, int K, int srcNK) {
  int i = blockIdx.x * 256 + threadIdx.x;
  if (i >= N * K) return;
  int n = i / K, k = i - n * K;
  float v = srcNK ? src[i] : src[(size_t)k * N + n];
  dst[i] = f32_to_bf16(v);
}

// ---------------------------------------------------------------------------
// GroupNorm (32 groups, C=320, HW=4096) -> token-major bf16 [4096,320]
// ---------------------------------------------------------------------------
__global__ void groupnorm_kernel(const float* __restrict__ x,
                                 const float* __restrict__ gw,
                                 const float* __restrict__ gb,
                                 ushort_t* __restrict__ xtok) {
  __shared__ float s1[256], s2[256];
  const int CPG = 10, HW = 4096;
  int g = blockIdx.x, tid = threadIdx.x;
  int c0 = g * CPG;
  float sum = 0.f, sq = 0.f;
  for (int i = tid; i < CPG * HW; i += 256) {
    float v = x[(size_t)(c0 + i / HW) * HW + (i % HW)];
    sum += v; sq += v * v;
  }
  s1[tid] = sum; s2[tid] = sq;
  __syncthreads();
  for (int s = 128; s > 0; s >>= 1) {
    if (tid < s) { s1[tid] += s1[tid + s]; s2[tid] += s2[tid + s]; }
    __syncthreads();
  }
  float mu = s1[0] / (CPG * HW);
  float var = s2[0] / (CPG * HW) - mu * mu;
  float rstd = rsqrtf(var + 1e-6f);
  for (int i = tid; i < CPG * HW; i += 256) {
    int c = c0 + i / HW, hw = i % HW;
    float v = (x[(size_t)c * HW + hw] - mu) * rstd * gw[c] + gb[c];
    xtok[(size_t)hw * 320 + c] = f32_to_bf16(v);
  }
}

// ---------------------------------------------------------------------------
// LayerNorm over D=320, one wave per row, bf16 out
// ---------------------------------------------------------------------------
__global__ void layernorm_kernel(const float* __restrict__ h,
                                 const float* __restrict__ w,
                                 const float* __restrict__ b,
                                 ushort_t* __restrict__ out, int M) {
  int lane = threadIdx.x & 31;
  int row = blockIdx.x * 8 + (threadIdx.x >> 5);
  if (row >= M) return;
  const float* r = h + (size_t)row * 320;
  float sum = 0.f, sq = 0.f;
  for (int i = lane; i < 320; i += 32) { float v = r[i]; sum += v; sq += v * v; }
#pragma unroll
  for (int o = 16; o > 0; o >>= 1) {
    sum += __shfl_xor(sum, o, 32);
    sq  += __shfl_xor(sq,  o, 32);
  }
  float mu = sum * (1.f / 320.f);
  float var = sq * (1.f / 320.f) - mu * mu;
  float rstd = rsqrtf(var + 1e-5f);
  ushort_t* o_ = out + (size_t)row * 320;
  for (int i = lane; i < 320; i += 32)
    o_[i] = f32_to_bf16((r[i] - mu) * rstd * w[i] + b[i]);
}

// ---------------------------------------------------------------------------
// bf16 WMMA GEMM, 32x32 wave tile (2x2 register blocking).
// C[M,N] = A[M,K] * Bt[N,K]^T; N must be a multiple of 32; M arbitrary
// (out-of-range A rows are clamped -> duplicated compute, never stored).
// Epilogue by mode:
//  1: +bias[n]   2: +res[m*N+n]   4: store f32 [m*N+n]   8: store bf16 [m*N+n]
// 16: transposed store: outF[n*M+m] = v + res[n*M+m]   (proj_out + x residual)
// ---------------------------------------------------------------------------
__global__ void gemm_bf16_kernel(const ushort_t* __restrict__ A,
                                 const ushort_t* __restrict__ Bt,
                                 const float* __restrict__ bias,
                                 const float* __restrict__ res,
                                 float* __restrict__ outF,
                                 ushort_t* __restrict__ outB,
                                 int M, int N, int K, int mode) {
  int wid  = (blockIdx.x * blockDim.x + threadIdx.x) >> 5;
  int lane = threadIdx.x & 31;
  int nt = N >> 5;               // 32-wide N tiles
  int mt = (M + 31) >> 5;        // 32-high M tiles
  if (wid >= mt * nt) return;
  int tm = wid / nt, tn = wid - tm * nt;
  int half = lane >> 4, l16 = lane & 15;

  // clamp rows so hot loop has no divergence; guarded at store time
  int r0 = tm * 32 + l16;      if (r0 > M - 1) r0 = M - 1;
  int r1 = tm * 32 + 16 + l16; if (r1 > M - 1) r1 = M - 1;
  const ushort_t* ap0 = A + (size_t)r0 * K;
  const ushort_t* ap1 = A + (size_t)r1 * K;
  const ushort_t* bp0 = Bt + (size_t)(tn * 32 + l16) * K + half * 16;
  const ushort_t* bp1 = Bt + (size_t)(tn * 32 + 16 + l16) * K + half * 16;

  v8f acc00 = {}, acc01 = {}, acc10 = {}, acc11 = {};
  for (int kb = 0; kb < K; kb += 32) {
    v16bf a0 = load_afrag(ap0 + kb, half);
    v16bf a1 = load_afrag(ap1 + kb, half);
    v16bf b0 = load_bfrag(bp0 + kb);
    v16bf b1 = load_bfrag(bp1 + kb);
    if (kb + 32 < K) {
      __builtin_prefetch(bp0 + kb + 32, 0, 3);  // global_prefetch_b8
      __builtin_prefetch(bp1 + kb + 32, 0, 3);
    }
    acc00 = wmma_bf16(a0, b0, acc00);
    acc01 = wmma_bf16(a0, b1, acc01);
    acc10 = wmma_bf16(a1, b0, acc10);
    acc11 = wmma_bf16(a1, b1, acc11);
  }

  v8f* accs[4] = { &acc00, &acc01, &acc10, &acc11 };
#pragma unroll
  for (int mi = 0; mi < 2; ++mi) {
#pragma unroll
    for (int ni = 0; ni < 2; ++ni) {
      float* cp = (float*)accs[mi * 2 + ni];
      int nn = tn * 32 + ni * 16 + l16;
      float bv = (mode & 1) ? bias[nn] : 0.0f;
#pragma unroll
      for (int r = 0; r < 8; ++r) {
        int m = tm * 32 + mi * 16 + r + half * 8;
        if (m >= M) continue;
        float v = cp[r] + bv;
        if (mode & 2)  v += res[(size_t)m * N + nn];
        if (mode & 4)  outF[(size_t)m * N + nn] = v;
        if (mode & 8)  outB[(size_t)m * N + nn] = f32_to_bf16(v);
        if (mode & 16) outF[(size_t)nn * M + m] = v + res[(size_t)nn * M + m];
      }
    }
  }
}

// ---------------------------------------------------------------------------
// Pack [Ntok,320] bf16 -> padded head layouts
//  Q/K:  [H=8][Np][64]  (d 40..63 zero; t >= Ntok zero; Q pre-scaled by DH^-0.5)
//  V^T:  [H=8][64][Nkvp] (d-major so attention B-fragments are contiguous)
// ---------------------------------------------------------------------------
__global__ void pack_qk_kernel(const ushort_t* __restrict__ in, ushort_t* __restrict__ out,
                               int Ntok, int Np, float scale) {
  int i = blockIdx.x * 256 + threadIdx.x;
  int total = 8 * Np * 64;
  if (i >= total) return;
  int d = i & 63;
  int t = (i >> 6) % Np;
  int h = i / (Np * 64);
  ushort_t v = 0;
  if (t < Ntok && d < 40)
    v = f32_to_bf16(bf16_to_f32(in[(size_t)t * 320 + h * 40 + d]) * scale);
  out[i] = v;
}

__global__ void pack_vt_kernel(const ushort_t* __restrict__ in, ushort_t* __restrict__ out,
                               int Ntok, int Nkvp) {
  int i = blockIdx.x * 256 + threadIdx.x;
  int total = 8 * 64 * Nkvp;
  if (i >= total) return;
  int t = i % Nkvp;
  int d = (i / Nkvp) & 63;
  int h = i / (Nkvp * 64);
  out[i] = (t < Ntok && d < 40) ? in[(size_t)t * 320 + h * 40 + d] : (ushort_t)0;
}

// ---------------------------------------------------------------------------
// Flash attention. One wave per (head, 16-query tile). Head dim padded to 64.
//  Q : [8][Nq][64] bf16 (pre-scaled)    Km : [8][Nkvp][64] bf16
//  Vt: [8][64][Nkvp] bf16               Out: [Nq][320] bf16
// S = Q*K^T via 2 chained 16x16x32 WMMAs per 16-key subtile; online softmax
// with half-wave shuffles; P transposed C->A layout through wave-private LDS
// (synced with s_wait_dscnt); O += P*V via 4 WMMAs.
// ---------------------------------------------------------------------------
__global__ void attn_kernel(const ushort_t* __restrict__ Q,
                            const ushort_t* __restrict__ Km,
                            const ushort_t* __restrict__ Vt,
                            ushort_t* __restrict__ Out,
                            int Nq, int Lkv, int Nkvp) {
  __shared__ __align__(16) ushort_t Pld[8][16 * 40];  // per-wave 16x32 P tile, stride 40
  int w    = threadIdx.x >> 5;
  int lane = threadIdx.x & 31;
  int gw   = blockIdx.x * 8 + w;
  int qtiles = Nq >> 4;
  if (gw >= qtiles * 8) return;
  int head = gw & 7;
  int qt   = gw >> 3;
  int half = lane >> 4, l16 = lane & 15;

  const ushort_t* qrow = Q + ((size_t)head * Nq + qt * 16 + l16) * 64;
  v16bf aq0 = load_afrag(qrow, half);        // K-dim (d) 0..31
  v16bf aq1 = load_afrag(qrow + 32, half);   // K-dim (d) 32..63

  v8f acc[4] = {};
  float rmax[8], rsum[8];
#pragma unroll
  for (int r = 0; r < 8; ++r) { rmax[r] = -3.0e38f; rsum[r] = 0.f; }
  ushort_t* pl = &Pld[w][0];

  for (int kb = 0; kb < Nkvp; kb += 32) {
    // ---- scores for 32 keys (two 16x16 C tiles) ----
    v8f S[2];
#pragma unroll
    for (int s = 0; s < 2; ++s) {
      int key = kb + s * 16 + l16;
      const ushort_t* kp = Km + ((size_t)head * Nkvp + key) * 64 + half * 16;
      v8f c = {};
      c = wmma_bf16(aq0, load_bfrag(kp), c);
      c = wmma_bf16(aq1, load_bfrag(kp + 32), c);
      if (key >= Lkv) {
        float* f = (float*)&c;
#pragma unroll
        for (int r = 0; r < 8; ++r) f[r] = -1.0e30f;
      }
      S[s] = c;
    }
    float* s0 = (float*)&S[0];
    float* s1 = (float*)&S[1];
    float corr[8];
    asm volatile("" ::: "memory");  // prior-iter LDS reads stay above these stores
#pragma unroll
    for (int r = 0; r < 8; ++r) {
      float mloc = fmaxf(s0[r], s1[r]);
#pragma unroll
      for (int o = 1; o < 16; o <<= 1) mloc = fmaxf(mloc, __shfl_xor(mloc, o, 32));
      float newm = fmaxf(rmax[r], mloc);
      float c_ = expf(rmax[r] - newm);
      float p0 = expf(s0[r] - newm);
      float p1 = expf(s1[r] - newm);
      float rs = p0 + p1;
#pragma unroll
      for (int o = 1; o < 16; o <<= 1) rs += __shfl_xor(rs, o, 32);
      rsum[r] = rsum[r] * c_ + rs;
      rmax[r] = newm;
      corr[r] = c_;
      int prow = r + half * 8;                 // C-layout row of this lane
      pl[prow * 40 + l16]      = f32_to_bf16(p0);
      pl[prow * 40 + 16 + l16] = f32_to_bf16(p1);
    }
#pragma unroll
    for (int s2 = 0; s2 < 4; ++s2) {
      float* ac = (float*)&acc[s2];
#pragma unroll
      for (int r = 0; r < 8; ++r) ac[r] *= corr[r];
    }
    asm volatile("s_wait_dscnt 0" ::: "memory");   // LDS store -> load (CDNA5 split counter)
    v16bf pa = load_afrag(pl + l16 * 40, half);    // P in A-fragment layout (16x32)
#pragma unroll
    for (int s2 = 0; s2 < 4; ++s2) {
      const ushort_t* vp =
          Vt + ((size_t)head * 64 + s2 * 16 + l16) * Nkvp + kb + half * 16;
      acc[s2] = wmma_bf16(pa, load_bfrag(vp), acc[s2]);
    }
    asm volatile("" ::: "memory");
  }

  float inv[8];
#pragma unroll
  for (int r = 0; r < 8; ++r) inv[r] = 1.0f / rsum[r];
#pragma unroll
  for (int s2 = 0; s2 < 4; ++s2) {
    int d = s2 * 16 + l16;
    if (d >= 40) continue;
    float* ac = (float*)&acc[s2];
#pragma unroll
    for (int r = 0; r < 8; ++r) {
      int m = qt * 16 + r + half * 8;
      Out[(size_t)m * 320 + head * 40 + d] = f32_to_bf16(ac[r] * inv[r]);
    }
  }
}

// ---------------------------------------------------------------------------
// GEGLU: in [4096,2560] bf16 -> out [4096,1280] bf16, exact gelu via erf
// ---------------------------------------------------------------------------
__global__ void geglu_kernel(const ushort_t* __restrict__ h, ushort_t* __restrict__ out) {
  int i = blockIdx.x * 256 + threadIdx.x;
  const int total = 4096 * 1280;
  if (i >= total) return;
  int j = i % 1280, t = i / 1280;
  float a = bf16_to_f32(h[(size_t)t * 2560 + j]);
  float g = bf16_to_f32(h[(size_t)t * 2560 + 1280 + j]);
  float ge = 0.5f * g * (1.0f + erff(g * 0.70710678118f));
  out[i] = f32_to_bf16(a * ge);
}

// ---------------------------------------------------------------------------
// Host orchestration
// ---------------------------------------------------------------------------
extern "C" void kernel_launch(void* const* d_in, const int* in_sizes, int n_in,
                              void* d_out, int out_size, void* d_ws, size_t ws_size,
                              hipStream_t stream) {
  (void)in_sizes; (void)n_in; (void)out_size; (void)ws_size;

  const int Nq = 4096, D = 320, CTX = 77, CTXP = 96, CD = 768, FFI = 1280, FF2 = 2560;
  const int H = 8, DHp = 64;
  const float QSCALE = 0.15811388300841897f;  // 40^-0.5

  const float* x       = (const float*)d_in[0];
  const float* context = (const float*)d_in[1];
  const float* gn_w = (const float*)d_in[2];
  const float* gn_b = (const float*)d_in[3];
  const float* pin_w = (const float*)d_in[4];
  const float* pin_b = (const float*)d_in[5];
  const float* ln1_w = (const float*)d_in[6];
  const float* ln1_b = (const float*)d_in[7];
  const float* q1w = (const float*)d_in[8];
  const float* k1w = (const float*)d_in[9];
  const float* v1w = (const float*)d_in[10];
  const float* o1w = (const float*)d_in[11];
  const float* o1b = (const float*)d_in[12];
  const float* ln2_w = (const float*)d_in[13];
  const float* ln2_b = (const float*)d_in[14];
  const float* q2w = (const float*)d_in[15];
  const float* k2w = (const float*)d_in[16];
  const float* v2w = (const float*)d_in[17];
  const float* o2w = (const float*)d_in[18];
  const float* o2b = (const float*)d_in[19];
  const float* ln3_w = (const float*)d_in[20];
  const float* ln3_b = (const float*)d_in[21];
  const float* ff1w = (const float*)d_in[22];
  const float* ff1b = (const float*)d_in[23];
  const float* ff2w = (const float*)d_in[24];
  const float* ff2b = (const float*)d_in[25];
  const float* poutw = (const float*)d_in[26];
  const float* poutb = (const float*)d_in[27];

  // ---- workspace bump allocator ----
  char* wsb = (char*)d_ws;
  size_t off = 0;
  auto alloc = [&](size_t bytes) -> void* {
    void* p = wsb + off;
    off += (bytes + 255) & ~(size_t)255;
    return p;
  };
  ushort_t* Xtok  = (ushort_t*)alloc((size_t)Nq * D * 2);
  ushort_t* Wpin  = (ushort_t*)alloc((size_t)D * D * 2);
  ushort_t* Wq1   = (ushort_t*)alloc((size_t)D * D * 2);
  ushort_t* Wk1   = (ushort_t*)alloc((size_t)D * D * 2);
  ushort_t* Wv1   = (ushort_t*)alloc((size_t)D * D * 2);
  ushort_t* Wo1   = (ushort_t*)alloc((size_t)D * D * 2);
  ushort_t* Wq2   = (ushort_t*)alloc((size_t)D * D * 2);
  ushort_t* Wk2   = (ushort_t*)alloc((size_t)D * CD * 2);
  ushort_t* Wv2   = (ushort_t*)alloc((size_t)D * CD * 2);
  ushort_t* Wo2   = (ushort_t*)alloc((size_t)D * D * 2);
  ushort_t* Wff1  = (ushort_t*)alloc((size_t)FF2 * D * 2);
  ushort_t* Wff2  = (ushort_t*)alloc((size_t)D * FFI * 2);
  ushort_t* Wpout = (ushort_t*)alloc((size_t)D * D * 2);
  ushort_t* Ctxb  = (ushort_t*)alloc((size_t)CTX * CD * 2);
  float*    hA    = (float*)alloc((size_t)Nq * D * 4);
  float*    hB    = (float*)alloc((size_t)Nq * D * 4);
  ushort_t* Lnb   = (ushort_t*)alloc((size_t)Nq * D * 2);
  ushort_t* Qb    = (ushort_t*)alloc((size_t)Nq * D * 2);
  ushort_t* Kb    = (ushort_t*)alloc((size_t)Nq * D * 2);
  ushort_t* Vb    = (ushort_t*)alloc((size_t)Nq * D * 2);
  ushort_t* Qp    = (ushort_t*)alloc((size_t)H * Nq * DHp * 2);
  ushort_t* Kp    = (ushort_t*)alloc((size_t)H * Nq * DHp * 2);
  ushort_t* Vtp   = (ushort_t*)alloc((size_t)H * DHp * Nq * 2);
  ushort_t* Attno = (ushort_t*)alloc((size_t)Nq * D * 2);
  ushort_t* Ffh   = (ushort_t*)alloc((size_t)Nq * FF2 * 2);
  ushort_t* Ffg   = (ushort_t*)alloc((size_t)Nq * FFI * 2);
  ushort_t* H3b   = (ushort_t*)alloc((size_t)Nq * D * 2);

  auto blocks = [](long long n) { return (int)((n + 255) / 256); };
  auto wconv = [&](const float* s, ushort_t* dst, int N, int K, int srcNK) {
    wconv_kernel<<<blocks((long long)N * K), 256, 0, stream>>>(s, dst, N, K, srcNK);
  };
  auto gemm = [&](const ushort_t* A, const ushort_t* Bt, const float* bias,
                  const float* res, float* oF, ushort_t* oB,
                  int M, int N, int K, int mode) {
    long long waves = (long long)((M + 31) / 32) * (N / 32);  // 32x32 wave tiles
    gemm_bf16_kernel<<<(int)((waves + 7) / 8), 256, 0, stream>>>(
        A, Bt, bias, res, oF, oB, M, N, K, mode);
  };

  // ---- weight / context conversion to bf16 Bt[N][K] ----
  wconv(pin_w, Wpin, D, D, 1);         // conv weight stored [out,in]
  wconv(q1w, Wq1, D, D, 0);            // attention weights stored (in,out)
  wconv(k1w, Wk1, D, D, 0);
  wconv(v1w, Wv1, D, D, 0);
  wconv(o1w, Wo1, D, D, 0);
  wconv(q2w, Wq2, D, D, 0);
  wconv(k2w, Wk2, D, CD, 0);
  wconv(v2w, Wv2, D, CD, 0);
  wconv(o2w, Wo2, D, D, 0);
  wconv(ff1w, Wff1, FF2, D, 0);
  wconv(ff2w, Wff2, D, FFI, 0);
  wconv(poutw, Wpout, D, D, 1);        // conv weight stored [out,in]
  wconv(context, Ctxb, 1, CTX * CD, 1);  // plain f32->bf16 flatten

  // ---- GroupNorm -> token-major bf16 ----
  groupnorm_kernel<<<32, 256, 0, stream>>>(x, gn_w, gn_b, Xtok);

  // ---- proj_in ----
  gemm(Xtok, Wpin, pin_b, nullptr, hA, nullptr, Nq, D, D, 1 | 4);

  // ---- self-attention block ----
  layernorm_kernel<<<Nq / 8, 256, 0, stream>>>(hA, ln1_w, ln1_b, Lnb, Nq);
  gemm(Lnb, Wq1, nullptr, nullptr, nullptr, Qb, Nq, D, D, 8);
  gemm(Lnb, Wk1, nullptr, nullptr, nullptr, Kb, Nq, D, D, 8);
  gemm(Lnb, Wv1, nullptr, nullptr, nullptr, Vb, Nq, D, D, 8);
  pack_qk_kernel<<<blocks((long long)H * Nq * DHp), 256, 0, stream>>>(Qb, Qp, Nq, Nq, QSCALE);
  pack_qk_kernel<<<blocks((long long)H * Nq * DHp), 256, 0, stream>>>(Kb, Kp, Nq, Nq, 1.0f);
  pack_vt_kernel<<<blocks((long long)H * DHp * Nq), 256, 0, stream>>>(Vb, Vtp, Nq, Nq);
  attn_kernel<<<(H * (Nq / 16)) / 8, 256, 0, stream>>>(Qp, Kp, Vtp, Attno, Nq, Nq, Nq);
  gemm(Attno, Wo1, o1b, hA, hB, nullptr, Nq, D, D, 1 | 2 | 4);

  // ---- cross-attention block ----
  layernorm_kernel<<<Nq / 8, 256, 0, stream>>>(hB, ln2_w, ln2_b, Lnb, Nq);
  gemm(Lnb, Wq2, nullptr, nullptr, nullptr, Qb, Nq, D, D, 8);
  gemm(Ctxb, Wk2, nullptr, nullptr, nullptr, Kb, CTX, D, CD, 8);
  gemm(Ctxb, Wv2, nullptr, nullptr, nullptr, Vb, CTX, D, CD, 8);
  pack_qk_kernel<<<blocks((long long)H * Nq * DHp), 256, 0, stream>>>(Qb, Qp, Nq, Nq, QSCALE);
  pack_qk_kernel<<<blocks((long long)H * CTXP * DHp), 256, 0, stream>>>(Kb, Kp, CTX, CTXP, 1.0f);
  pack_vt_kernel<<<blocks((long long)H * DHp * CTXP), 256, 0, stream>>>(Vb, Vtp, CTX, CTXP);
  attn_kernel<<<(H * (Nq / 16)) / 8, 256, 0, stream>>>(Qp, Kp, Vtp, Attno, Nq, CTX, CTXP);
  gemm(Attno, Wo2, o2b, hB, hA, nullptr, Nq, D, D, 1 | 2 | 4);

  // ---- GEGLU feed-forward ----
  layernorm_kernel<<<Nq / 8, 256, 0, stream>>>(hA, ln3_w, ln3_b, Lnb, Nq);
  gemm(Lnb, Wff1, ff1b, nullptr, nullptr, Ffh, Nq, FF2, D, 1 | 8);
  geglu_kernel<<<blocks((long long)Nq * FFI), 256, 0, stream>>>(Ffh, Ffg);
  gemm(Ffg, Wff2, ff2b, hA, hB, nullptr, Nq, D, FFI, 1 | 2 | 4);

  // ---- proj_out + input residual (transposed store back to [C,H,W]) ----
  wconv(hB, H3b, 1, Nq * D, 1);  // f32 -> bf16 of the token stream
  gemm(H3b, Wpout, poutb, x, (float*)d_out, nullptr, Nq, D, D, 1 | 16);
}
